// Block_35012573397344
// MI455X (gfx1250) — matrix operand order repaired and checked by reading, not compiled, not run on hardware
//
#include <hip/hip_runtime.h>
#include <hip/hip_bf16.h>
#include <math.h>

// ---------------------------------------------------------------------------
// Types for CDNA5 WMMA (wave32): v_wmma_f32_16x16x32_bf16
// ---------------------------------------------------------------------------
typedef __bf16 bf16_t;
typedef __attribute__((ext_vector_type(4)))  __bf16 bf16x4;
typedef __attribute__((ext_vector_type(8)))  __bf16 bf16x8;
typedef __attribute__((ext_vector_type(16))) __bf16 bf16x16;
typedef __attribute__((ext_vector_type(8)))  float  f32x8;
typedef __attribute__((ext_vector_type(4)))  float  f32x4;

union Frag {
    bf16x16 v;
    bf16x8  h[2];
};

// Problem constants (match reference)
#define Bb  2
#define Tt  2048
#define Cc  1024
#define Hh  16
#define HS  64
#define Mm  (Bb * Tt)     // 4096 rows
#define LN_EPS 1e-5f

static __device__ __forceinline__ f32x8 wmma_bf16(const Frag& a, const Frag& b, f32x8 c) {
    return __builtin_amdgcn_wmma_f32_16x16x32_bf16(
        /*neg_a=*/false, a.v, /*neg_b=*/false, b.v,
        /*c_mod=*/(short)0, c, /*reuse_a=*/false, /*reuse_b=*/false);
}

// ---------------------------------------------------------------------------
// CDNA5 async global->LDS copy (ASYNCcnt-tracked), per CDNA5 ISA
// LDS address = low 32 bits of the generic pointer (flat aperture truncation).
// ---------------------------------------------------------------------------
static __device__ __forceinline__ uint32_t lds_addr32(const void* p) {
    return (uint32_t)(uintptr_t)p;
}
static __device__ __forceinline__ void async_copy_b128(uint32_t lds_addr, const void* gptr) {
    asm volatile("global_load_async_to_lds_b128 %0, %1, off"
                 :: "v"(lds_addr), "v"(gptr)
                 : "memory");
}
static __device__ __forceinline__ void wait_async0() {
    asm volatile("s_wait_asynccnt 0x0" ::: "memory");
}

// ---------------------------------------------------------------------------
// Weight packing: fp32 [K][N] -> bf16 transposed [N][K] (B-fragment friendly)
// ---------------------------------------------------------------------------
__global__ __launch_bounds__(256)
void transpose_pack_kernel(const float* __restrict__ in, bf16_t* __restrict__ out,
                           int K, int N) {
    size_t i = (size_t)blockIdx.x * 256 + threadIdx.x;   // coalesced read order
    if (i >= (size_t)K * N) return;
    int n = (int)(i % N);
    int k = (int)(i / N);
    out[(size_t)n * K + k] = (bf16_t)in[i];
}

// wq/wk/wv: fp32 [H][C][HS] -> bf16 [H*HS rows n][C], n = head*HS + d
__global__ __launch_bounds__(256)
void pack_qkv_kernel(const float* __restrict__ w, bf16_t* __restrict__ out) {
    size_t i = (size_t)blockIdx.x * 256 + threadIdx.x;   // over 1024*1024
    int n = (int)(i >> 10);
    int c = (int)(i & 1023);
    int h = n >> 6;
    int d = n & 63;
    out[(size_t)n * Cc + c] = (bf16_t)w[((size_t)h * Cc + c) * HS + d];
}

// ---------------------------------------------------------------------------
// LayerNorm: fp32 row [1024] -> bf16 row
// ---------------------------------------------------------------------------
__global__ __launch_bounds__(256)
void ln_bf16_kernel(const float* __restrict__ x,
                    const float* __restrict__ gamma,
                    const float* __restrict__ beta,
                    bf16_t* __restrict__ out) {
    const int row = blockIdx.x;
    const int tid = threadIdx.x;
    const float* xr = x + (size_t)row * Cc;
    f32x4 v = *(const f32x4*)(xr + tid * 4);

    float s  = v[0] + v[1] + v[2] + v[3];
    float sq = v[0]*v[0] + v[1]*v[1] + v[2]*v[2] + v[3]*v[3];

    __shared__ float rs[256];
    __shared__ float rq[256];
    rs[tid] = s; rq[tid] = sq;
    __syncthreads();
    #pragma unroll
    for (int off = 128; off > 0; off >>= 1) {
        if (tid < off) { rs[tid] += rs[tid + off]; rq[tid] += rq[tid + off]; }
        __syncthreads();
    }
    float mean = rs[0] * (1.0f / Cc);
    float var  = rq[0] * (1.0f / Cc) - mean * mean;
    float rstd = rsqrtf(var + LN_EPS);

    const int c = tid * 4;
    bf16x4 o;
    #pragma unroll
    for (int k = 0; k < 4; ++k)
        o[k] = (bf16_t)((v[k] - mean) * rstd * gamma[c + k] + beta[c + k]);
    *(bf16x4*)(out + (size_t)row * Cc + c) = o;
}

// ---------------------------------------------------------------------------
// bf16 GEMM: D[M,N] = A[M,K] * Bt[N,K]^T  (+epilogue)
// Block tile 128x128 (4 waves x 64x64 wave tiles), K-step 32.
// A/B tiles double-buffered in LDS, filled with async global->LDS copies.
// K-loop unrolled x2 so LDS buffer addresses are compile-time constants.
// ---------------------------------------------------------------------------
constexpr int EPI_QKV       = 0;   // scatter bf16 into qkv[3][B][H][T][HS]
constexpr int EPI_RES_F32   = 1;   // f32 out = acc + bias[n] + resid[m*N+n]
constexpr int EPI_RELU_BF16 = 2;   // bf16 out = relu(acc + bias[n])

template<int EPI>
__global__ __launch_bounds__(128, 2)
void gemm_bf16_kernel(const bf16_t* __restrict__ A,
                      const bf16_t* __restrict__ Bt,
                      const float*  __restrict__ bias,
                      const float*  __restrict__ resid,
                      float*        __restrict__ outF,
                      bf16_t*       __restrict__ outB,
                      int M, int N, int K) {
    __shared__ __align__(16) bf16_t Ash[2][128 * 32];   // 2 x 8KB
    __shared__ __align__(16) bf16_t Bsh[2][128 * 32];   // 2 x 8KB

    const int tid  = threadIdx.x;
    const int lane = tid & 31;
    const int wave = tid >> 5;
    const int g    = lane >> 4;     // lane-half (K-group select)
    const int lr   = lane & 15;     // A row / B,C column within 16-tile

    const int bm = blockIdx.y * 128;
    const int bn = blockIdx.x * 128;
    const int wm = (wave >> 1) * 64;
    const int wn = (wave & 1) * 64;

    f32x8 acc[4][4];
    #pragma unroll
    for (int i = 0; i < 4; ++i)
        #pragma unroll
        for (int j = 0; j < 4; ++j)
            acc[i][j] = (f32x8)0.0f;

    const uint32_t a_lds = lds_addr32(&Ash[0][0]);
    const uint32_t b_lds = lds_addr32(&Bsh[0][0]);

    // Issue one 128x32 bf16 tile of A and B into LDS buffer `buf` (async).
    // 512 chunks of 8 bf16 per matrix; 128 threads -> 4 chunks each.
    auto issue_tile = [&](int buf, int k0) {
        #pragma unroll
        for (int c = 0; c < 4; ++c) {
            const int chunk = tid + c * 128;
            const int row   = chunk >> 2;          // 0..127
            const int col   = (chunk & 3) * 8;     // 0,8,16,24
            const uint32_t loff = (uint32_t)(buf * 8192 + (row * 32 + col) * 2);
            async_copy_b128(a_lds + loff, A  + (size_t)(bm + row) * K + k0 + col);
            async_copy_b128(b_lds + loff, Bt + (size_t)(bn + row) * K + k0 + col);
        }
    };

    // MAC one staged 128x32 tile pair into the accumulators.
    auto compute_tile = [&](const bf16_t* Abuf, const bf16_t* Bbuf) {
        Frag a[4], b[4];
        #pragma unroll
        for (int i = 0; i < 4; ++i) {
            const bf16_t* ap = Abuf + (wm + 16 * i + lr) * 32;
            a[i].h[0] = *(const bf16x8*)(ap + 8 * g);
            a[i].h[1] = *(const bf16x8*)(ap + 16 + 8 * g);
        }
        #pragma unroll
        for (int j = 0; j < 4; ++j) {
            const bf16_t* bp = Bbuf + (wn + 16 * j + lr) * 32;
            b[j].h[0] = *(const bf16x8*)(bp + 8 * g);
            b[j].h[1] = *(const bf16x8*)(bp + 16 + 8 * g);
        }
        #pragma unroll
        for (int i = 0; i < 4; ++i)
            #pragma unroll
            for (int j = 0; j < 4; ++j)
                acc[i][j] = wmma_bf16(a[i], b[j], acc[i][j]);
    };

    issue_tile(0, 0);
    const int nk = K >> 5;                 // even for K in {1024, 4096}
    for (int kt = 0; kt < nk; kt += 2) {
        // stage 0: consume buffer 0, prefetch into buffer 1
        wait_async0();
        __syncthreads();
        if (kt + 1 < nk) issue_tile(1, (kt + 1) * 32);
        compute_tile(&Ash[0][0], &Bsh[0][0]);
        // stage 1: consume buffer 1, prefetch into buffer 0
        wait_async0();
        __syncthreads();
        if (kt + 2 < nk) issue_tile(0, (kt + 2) * 32);
        compute_tile(&Ash[1][0], &Bsh[1][0]);
    }

    const int m0 = bm + wm;
    const int n0 = bn + wn;
    const int basem = 8 * g;
    #pragma unroll
    for (int i = 0; i < 4; ++i) {
        #pragma unroll
        for (int j = 0; j < 4; ++j) {
            const int n = n0 + 16 * j + lr;
            #pragma unroll
            for (int r = 0; r < 8; ++r) {
                const int m = m0 + 16 * i + basem + r;
                float v = acc[i][j][r];
                if constexpr (EPI == EPI_QKV) {
                    const int sel  = n >> 10;       // 0=q 1=k 2=v (N==3072)
                    const int nc   = n & 1023;
                    const int head = nc >> 6;
                    const int d    = nc & 63;
                    const int bix  = m >> 11;       // m / T
                    const int t    = m & 2047;      // m % T
                    size_t idx = ((((size_t)sel * Bb + bix) * Hh + head) * Tt + t) * HS + d;
                    outB[idx] = (bf16_t)v;
                } else if constexpr (EPI == EPI_RES_F32) {
                    size_t idx = (size_t)m * N + n;
                    outF[idx] = v + bias[n] + resid[idx];
                } else {  // EPI_RELU_BF16
                    size_t idx = (size_t)m * N + n;
                    float t = v + bias[n];
                    outB[idx] = (bf16_t)(t > 0.0f ? t : 0.0f);
                }
            }
        }
    }
}

// ---------------------------------------------------------------------------
// Flash-style causal attention. Grid: (T/64, B*H), block 128 (4 waves).
// Wave w handles 16 query rows; block tile = 64 queries x 64 keys per step.
// scale = C^-0.5 = 1/32 (matches reference quirk).
// ---------------------------------------------------------------------------
__global__ __launch_bounds__(128, 1)
void attn_kernel(const bf16_t* __restrict__ qkv,   // [3][B][H][T][HS]
                 bf16_t* __restrict__ attn_out) {  // [B*T][C]
    __shared__ __align__(16) bf16_t Ks[64 * 64];       // K tile [s][d]
    __shared__ __align__(16) bf16_t Vt[64 * 64];       // V tile transposed [d][s]
    __shared__ __align__(16) bf16_t Ps[4][16 * 64];    // per-wave P tile [m][s]

    const int tq   = blockIdx.x * 64;
    const int bh   = blockIdx.y;
    const int bix  = bh / Hh;
    const int head = bh % Hh;

    const size_t plane = (size_t)Tt * HS;
    const bf16_t* Q  = qkv + (((size_t)0 * Bb + bix) * Hh + head) * plane;
    const bf16_t* Kg = qkv + (((size_t)1 * Bb + bix) * Hh + head) * plane;
    const bf16_t* Vg = qkv + (((size_t)2 * Bb + bix) * Hh + head) * plane;

    const int tid  = threadIdx.x;
    const int lane = tid & 31;
    const int wave = tid >> 5;
    const int g    = lane >> 4;
    const int lr   = lane & 15;

    const uint32_t ks_lds = lds_addr32(&Ks[0]);

    // Preload Q A-fragments for this wave's 16 rows (d = 0..63 -> 2 K-chunks)
    Frag qa[2];
    const bf16_t* qrow = Q + (size_t)(tq + wave * 16 + lr) * HS;
    #pragma unroll
    for (int c = 0; c < 2; ++c) {
        qa[c].h[0] = *(const bf16x8*)(qrow + c * 32 + 8 * g);
        qa[c].h[1] = *(const bf16x8*)(qrow + c * 32 + 16 + 8 * g);
    }

    f32x8 oacc[4];
    #pragma unroll
    for (int j = 0; j < 4; ++j) oacc[j] = (f32x8)0.0f;
    float mrun[8], lrun[8];
    #pragma unroll
    for (int r = 0; r < 8; ++r) { mrun[r] = -INFINITY; lrun[r] = 0.0f; }

    const float scale = 0.03125f;       // 1024^-0.5
    const int   nst   = (tq >> 6) + 1;  // causal: s-tiles 0..tq

    for (int st = 0; st < nst; ++st) {
        const int s0 = st * 64;
        __syncthreads();   // previous iteration's LDS reads done

        // K tile via async global->LDS (16B chunks).
        #pragma unroll
        for (int c = 0; c < 4; ++c) {
            int chunk = tid + c * 128;          // 512 chunks of 8 bf16
            int srow  = chunk >> 3;
            int dcol  = (chunk & 7) * 8;
            async_copy_b128(ks_lds + (uint32_t)((srow * 64 + dcol) * 2),
                            Kg + (size_t)(s0 + srow) * HS + dcol);
        }
        // V tile: vectorized 16B global loads, transposed scatter into LDS.
        #pragma unroll
        for (int c = 0; c < 4; ++c) {
            int chunk = tid + c * 128;          // 512 chunks of 8 bf16
            int srow  = chunk >> 3;
            int dcol  = (chunk & 7) * 8;
            bf16x8 vv = *(const bf16x8*)(Vg + (size_t)(s0 + srow) * HS + dcol);
            #pragma unroll
            for (int e = 0; e < 8; ++e)
                Vt[(dcol + e) * 64 + srow] = vv[e];
        }
        wait_async0();
        __syncthreads();

        // S = Q * K^T  (16 x 64 per wave)
        f32x8 sacc[4];
        #pragma unroll
        for (int j = 0; j < 4; ++j) sacc[j] = (f32x8)0.0f;
        #pragma unroll
        for (int c = 0; c < 2; ++c) {
            #pragma unroll
            for (int j = 0; j < 4; ++j) {
                Frag kb;
                const bf16_t* kp = Ks + (16 * j + lr) * 64 + c * 32 + 8 * g;
                kb.h[0] = *(const bf16x8*)(kp);
                kb.h[1] = *(const bf16x8*)(kp + 16);
                sacc[j] = wmma_bf16(qa[c], kb, sacc[j]);
            }
        }

        // Scale + causal mask (only the diagonal tile needs masking)
        const bool diag = (s0 == tq);
        #pragma unroll
        for (int j = 0; j < 4; ++j) {
            const int s = s0 + 16 * j + lr;
            #pragma unroll
            for (int r = 0; r < 8; ++r) {
                const int rowq = tq + wave * 16 + 8 * g + r;
                float val = sacc[j][r] * scale;
                if (diag && s > rowq) val = -INFINITY;
                sacc[j][r] = val;
            }
        }

        // Online softmax (row stats across the 16-lane half via butterfly)
        #pragma unroll
        for (int r = 0; r < 8; ++r) {
            float t = fmaxf(fmaxf(sacc[0][r], sacc[1][r]),
                            fmaxf(sacc[2][r], sacc[3][r]));
            t = fmaxf(t, __shfl_xor(t, 1, 32));
            t = fmaxf(t, __shfl_xor(t, 2, 32));
            t = fmaxf(t, __shfl_xor(t, 4, 32));
            t = fmaxf(t, __shfl_xor(t, 8, 32));
            const float mnew = fmaxf(mrun[r], t);
            const float corr = __expf(mrun[r] - mnew);
            mrun[r] = mnew;
            float psum = 0.0f;
            #pragma unroll
            for (int j = 0; j < 4; ++j) {
                float p = __expf(sacc[j][r] - mnew);
                sacc[j][r] = p;
                psum += p;
            }
            psum += __shfl_xor(psum, 1, 32);
            psum += __shfl_xor(psum, 2, 32);
            psum += __shfl_xor(psum, 4, 32);
            psum += __shfl_xor(psum, 8, 32);
            lrun[r] = lrun[r] * corr + psum;
            #pragma unroll
            for (int j = 0; j < 4; ++j) oacc[j][r] *= corr;
        }

        // P (C-layout) -> LDS -> A-fragment layout
        bf16_t* pw = &Ps[wave][0];
        #pragma unroll
        for (int j = 0; j < 4; ++j)
            #pragma unroll
            for (int r = 0; r < 8; ++r)
                pw[(8 * g + r) * 64 + 16 * j + lr] = (bf16_t)sacc[j][r];
        __syncthreads();

        // O += P * V   (A = P[16 x 64s], B = Vt[d][s])
        #pragma unroll
        for (int c = 0; c < 2; ++c) {
            Frag pa;
            const bf16_t* pr = pw + lr * 64 + c * 32;
            pa.h[0] = *(const bf16x8*)(pr + 8 * g);
            pa.h[1] = *(const bf16x8*)(pr + 16 + 8 * g);
            #pragma unroll
            for (int jd = 0; jd < 4; ++jd) {
                Frag vb;
                const bf16_t* vp = Vt + (16 * jd + lr) * 64 + c * 32 + 8 * g;
                vb.h[0] = *(const bf16x8*)(vp);
                vb.h[1] = *(const bf16x8*)(vp + 16);
                oacc[jd] = wmma_bf16(pa, vb, oacc[jd]);
            }
        }
    }

    // Normalize and store (head concat into [B*T][C])
    #pragma unroll
    for (int jd = 0; jd < 4; ++jd) {
        const int d = 16 * jd + lr;
        #pragma unroll
        for (int r = 0; r < 8; ++r) {
            const int rowq = tq + wave * 16 + 8 * g + r;
            const float o = oacc[jd][r] / lrun[r];
            attn_out[((size_t)bix * Tt + rowq) * Cc + head * HS + d] = (bf16_t)o;
        }
    }
}

// ---------------------------------------------------------------------------
// Host-side orchestration
// ---------------------------------------------------------------------------
extern "C" void kernel_launch(void* const* d_in, const int* in_sizes, int n_in,
                              void* d_out, int out_size, void* d_ws, size_t ws_size,
                              hipStream_t stream) {
    const float* x      = (const float*)d_in[0];
    const float* wq     = (const float*)d_in[1];
    const float* wk     = (const float*)d_in[2];
    const float* wv     = (const float*)d_in[3];
    const float* w_proj = (const float*)d_in[4];
    const float* b_proj = (const float*)d_in[5];
    const float* w1     = (const float*)d_in[6];
    const float* b1     = (const float*)d_in[7];
    const float* w2     = (const float*)d_in[8];
    const float* b2     = (const float*)d_in[9];
    const float* ln1_g  = (const float*)d_in[10];
    const float* ln1_b  = (const float*)d_in[11];
    const float* ln2_g  = (const float*)d_in[12];
    const float* ln2_b  = (const float*)d_in[13];

    char* ws = (char*)d_ws;
    // bf16 weight copies (transposed [N][K]) + bf16 activations + f32 residual
    bf16_t* wqkvt  = (bf16_t*)(ws + 0);               //  6 MB: [3C][C]
    bf16_t* wprojt = (bf16_t*)(ws + 6291456);         //  2 MB: [C][C]
    bf16_t* w1t    = (bf16_t*)(ws + 8388608);         //  8 MB: [4C][C]
    bf16_t* w2t    = (bf16_t*)(ws + 16777216);        //  8 MB: [C][4C]
    bf16_t* hbuf   = (bf16_t*)(ws + 25165824);        //  8 MB: ln1(x)
    bf16_t* qkvb   = (bf16_t*)(ws + 33554432);        // 24 MB: [3][B][H][T][HS]
    bf16_t* attnb  = (bf16_t*)(ws + 58720256);        //  8 MB: [B*T][C]
    float*  x1     = (float* )(ws + 67108864);        // 16 MB: x + proj
    bf16_t* h2     = (bf16_t*)(ws + 83886080);        //  8 MB: ln2(x1)
    bf16_t* ff     = (bf16_t*)(ws + 92274688);        // 32 MB: [B*T][4C]

    // 1) Pack weights to bf16 (transposed)
    pack_qkv_kernel<<<4096, 256, 0, stream>>>(wq, wqkvt);
    pack_qkv_kernel<<<4096, 256, 0, stream>>>(wk, wqkvt + (size_t)Cc * Cc);
    pack_qkv_kernel<<<4096, 256, 0, stream>>>(wv, wqkvt + 2 * (size_t)Cc * Cc);
    transpose_pack_kernel<<<4096,  256, 0, stream>>>(w_proj, wprojt, Cc, Cc);
    transpose_pack_kernel<<<16384, 256, 0, stream>>>(w1, w1t, Cc, 4 * Cc);
    transpose_pack_kernel<<<16384, 256, 0, stream>>>(w2, w2t, 4 * Cc, Cc);

    // 2) LN1
    ln_bf16_kernel<<<Mm, 256, 0, stream>>>(x, ln1_g, ln1_b, hbuf);

    // 3) QKV projection: [4096 x 3072 x 1024]
    gemm_bf16_kernel<EPI_QKV><<<dim3(24, 32), 128, 0, stream>>>(
        hbuf, wqkvt, nullptr, nullptr, nullptr, qkvb, Mm, 3 * Cc, Cc);

    // 4) Causal attention
    attn_kernel<<<dim3(Tt / 64, Bb * Hh), 128, 0, stream>>>(qkvb, attnb);

    // 5) Output projection + residual: x1 = x + attn @ w_proj + b_proj
    gemm_bf16_kernel<EPI_RES_F32><<<dim3(8, 32), 128, 0, stream>>>(
        attnb, wprojt, b_proj, x, x1, nullptr, Mm, Cc, Cc);

    // 6) LN2
    ln_bf16_kernel<<<Mm, 256, 0, stream>>>(x1, ln2_g, ln2_b, h2);

    // 7) MLP up + ReLU: [4096 x 4096 x 1024]
    gemm_bf16_kernel<EPI_RELU_BF16><<<dim3(32, 32), 128, 0, stream>>>(
        h2, w1t, b1, nullptr, nullptr, ff, Mm, 4 * Cc, Cc);

    // 8) MLP down + residual into d_out: [4096 x 1024 x 4096]
    gemm_bf16_kernel<EPI_RES_F32><<<dim3(8, 32), 128, 0, stream>>>(
        ff, w2t, b2, x1, (float*)d_out, nullptr, Mm, Cc, 4 * Cc);
}
